// GAT_HotpotQA_66649302499991
// MI455X (gfx1250) — matrix operand order repaired and checked by reading, not compiled, not run on hardware
//
#include <hip/hip_runtime.h>
#include <hip/hip_fp16.h>
#include <math.h>

#define BB 128
#define NN 200
#define NP 208   // padded rows (13 * 16)
#define FF 768
#define FO 32
#define HH 8
#define HO 256   // HH*FO
#define NEGV (-9e9f)
#define BNS 0.9999950000374997f   // 1/sqrt(1+1e-5)
#define ALPHA_ 0.3f

typedef __attribute__((ext_vector_type(16))) _Float16 v16h;
typedef __attribute__((ext_vector_type(8)))  _Float16 v8h;
typedef __attribute__((ext_vector_type(8)))  float    v8f;

static __device__ __forceinline__ v8f wmma_f16(v16h a, v16h b, v8f c) {
  return __builtin_amdgcn_wmma_f32_16x16x32_f16(false, a, false, b, (short)0, c, false, false);
}

// A fragment (16x32 f16): lane half hh: VGPR0-3 hold K = hh*8+0..7, VGPR4-7 hold K = 16+hh*8+0..7
static __device__ __forceinline__ v16h ld_a_frag(const _Float16* rowp, int hh) {
  union { v16h v; v8h h[2]; } u;
  u.h[0] = *(const v8h*)(rowp + hh * 8);
  u.h[1] = *(const v8h*)(rowp + 16 + hh * 8);
  return u.v;
}
// B fragment (32x16 f16): per column (lane&15), lane half hh: VGPR0-3 hold K = hh*8+0..7, VGPR4-7 hold K = 16+hh*8+0..7
static __device__ __forceinline__ v16h ld_b_frag(const _Float16* colp, int hh) {
  union { v16h v; v8h h[2]; } u;
  u.h[0] = *(const v8h*)(colp + hh * 8);
  u.h[1] = *(const v8h*)(colp + 16 + hh * 8);
  return u.v;
}

static __device__ __forceinline__ float leakyf(float x) { return x > 0.f ? x : ALPHA_ * x; }
static __device__ __forceinline__ float eluf(float x)   { return x > 0.f ? x : expm1f(x); }

// ---------------- kernel 0a: pack W_heads [8,768,32] -> WallT f16 [256][768] ----------------
__global__ void k0_prep_w(const float* __restrict__ Wheads, _Float16* __restrict__ wallT) {
  int idx = blockIdx.x * 256 + threadIdx.x;
  if (idx >= HO * FF) return;
  int c = idx / FF, k = idx - c * FF;
  wallT[c * FF + k] = (_Float16)Wheads[((c >> 5) * FF + k) * FO + (c & 31)];
}

// ---------------- kernel 0b: feat f32 [B,200,768] -> featH f16 [B,208,768], pad rows zero ----------
__global__ void k0_prep_x(const float* __restrict__ feat, _Float16* __restrict__ featH) {
  int idx = blockIdx.x * 256 + threadIdx.x;          // one thread per 8 elements
  if (idx >= BB * NP * FF / 8) return;
  int e = idx * 8;
  int b = e / (NP * FF);
  int rem = e - b * NP * FF;
  int row = rem / FF;
  int k = rem - row * FF;
  v8h o;
  if (row < NN) {
    const float* p = feat + ((size_t)b * NN + row) * FF + k;
    float4 f0 = *(const float4*)p;
    float4 f1 = *(const float4*)(p + 4);
    o[0] = (_Float16)f0.x; o[1] = (_Float16)f0.y; o[2] = (_Float16)f0.z; o[3] = (_Float16)f0.w;
    o[4] = (_Float16)f1.x; o[5] = (_Float16)f1.y; o[6] = (_Float16)f1.z; o[7] = (_Float16)f1.w;
  } else {
    o = (v8h)(_Float16)0.0f;
  }
  *(v8h*)(featH + (size_t)e) = o;
}

// ---------------- kernel 1: Hall[b,n,256] = featH[b] (208x768) @ WallT^T, f16 WMMA ----------------
// No LDS, no barriers: each lane streams its A row (padded f16) and B columns directly.
__global__ void k1_gemm(const _Float16* __restrict__ featH, const _Float16* __restrict__ wallT,
                        float* __restrict__ hall) {
  const int b = blockIdx.y, rt = blockIdx.x;
  const int row0 = rt * 16;
  const int tid = threadIdx.x;
  const int wid = tid >> 5, lane = tid & 31;
  const int m = lane & 15, hh = lane >> 4;
  const int colbase = wid * 32;
  v8f acc0 = {0, 0, 0, 0, 0, 0, 0, 0}, acc1 = acc0;
  const _Float16* arow  = featH + ((size_t)b * NP + row0 + m) * FF;
  const _Float16* bcol0 = wallT + (size_t)(colbase + m) * FF;
  const _Float16* bcol1 = wallT + (size_t)(colbase + 16 + m) * FF;
#pragma unroll 4
  for (int kk = 0; kk < FF; kk += 32) {
    v16h af = ld_a_frag(arow + kk, hh);
    v16h b0 = ld_b_frag(bcol0 + kk, hh);
    v16h b1 = ld_b_frag(bcol1 + kk, hh);
    acc0 = wmma_f16(af, b0, acc0);
    acc1 = wmma_f16(af, b1, acc1);
  }
  for (int r = 0; r < 8; ++r) {
    int row = row0 + r + 8 * hh;   // D layout: lanes 16-31 hold M = 8 + r
    if (row < NN) {
      hall[((size_t)b * NN + row) * HO + colbase + m]      = acc0[r];
      hall[((size_t)b * NN + row) * HO + colbase + 16 + m] = acc1[r];
    }
  }
}

// ---------------- kernel 2: head attention scores s1/s2 [H,B,N] ----------------
__global__ void k2_scores(const float* __restrict__ hall, const float* __restrict__ aheads,
                          float* __restrict__ s1, float* __restrict__ s2) {
  int idx = blockIdx.x * 256 + threadIdx.x;
  if (idx >= HH * BB * NN) return;
  int h = idx / (BB * NN);
  int r = idx - h * BB * NN;
  int b = r / NN, n = r - b * NN;
  const float* hp = hall + ((size_t)b * NN + n) * HO + h * FO;
  const float* a  = aheads + h * 2 * FO;
  float x1 = 0.f, x2 = 0.f;
  for (int fo = 0; fo < FO; ++fo) { float hv = hp[fo]; x1 += hv * a[fo]; x2 += hv * a[FO + fo]; }
  s1[idx] = x1;
  s2[idx] = x2;
}

// ---------------- kernel 3: per-(b,h) masked softmax attention, hp = att @ h via WMMA ----------------
__global__ void k3_attn(const float* __restrict__ hall, const float* __restrict__ s1g,
                        const float* __restrict__ s2g, const int* __restrict__ adj,
                        const float* __restrict__ bnh_g, const float* __restrict__ bnh_b,
                        const float* __restrict__ bnt_g, const float* __restrict__ bnt_b,
                        float* __restrict__ xcat) {
  __shared__ __align__(16) _Float16 hT[32][208];       // transposed head tile, K-contiguous
  __shared__ __align__(16) _Float16 attW[8][16][32];   // per-wave P chunk
  __shared__ float s1L[NN], s2L[NN];
  __shared__ float invdW[8][16];
  const int h = blockIdx.x, b = blockIdx.y;
  const int tid = threadIdx.x;

  for (int e = tid; e < 32 * 208; e += 256) {
    int n = e / 208, j = e - n * 208;
    float v = (j < NN) ? hall[((size_t)b * NN + j) * HO + h * FO + n] : 0.f;
    hT[n][j] = (_Float16)v;
  }
  const float* s1p = s1g + ((size_t)h * BB + b) * NN;
  const float* s2p = s2g + ((size_t)h * BB + b) * NN;
  for (int j = tid; j < NN; j += 256) { s1L[j] = s1p[j]; s2L[j] = s2p[j]; }
  __syncthreads();

  const int wid = tid >> 5, lane = tid & 31;
  const int r = lane & 15, sub = lane >> 4;
  const int* adjB = adj + (size_t)b * NN * NN;

  for (int rt = wid; rt < 13; rt += 8) {
    const int row0 = rt * 16;
    const int row = row0 + r;
    const bool vrow = row < NN;
    const int* adjR = adjB + row * NN;
    const float s1v = vrow ? s1L[row] : 0.f;

    // pass A: per-row masked max (two lanes per row, j parity split)
    float mx = -3.0e38f;
    if (vrow) {
      for (int t = 0; t < 100; ++t) {
        int j = 2 * t + sub;
        float e  = leakyf(s1v + s2L[j]);
        float em = (adjR[j] > 0) ? e : NEGV;
        mx = fmaxf(mx, em);
      }
    }
    mx = fmaxf(mx, __shfl_xor(mx, 16, 32));

    // pass B: stream exp-numerator chunks into attW (same-wave LDS, DS in-order), WMMA accumulate
    v8f acc0 = {0, 0, 0, 0, 0, 0, 0, 0}, acc1 = acc0;
    float psum = 0.f;
    for (int kc = 0; kc < 7; ++kc) {
      v8h lo, hi;
#pragma unroll
      for (int c = 0; c < 16; ++c) {
        int j = kc * 32 + sub * 16 + c;
        float p = 0.f;
        if (vrow && j < NN) {
          float e  = leakyf(s1v + s2L[j]);
          float em = (adjR[j] > 0) ? e : NEGV;
          p = __expf(em - mx);
        }
        psum += p;
        if (c < 8) lo[c] = (_Float16)p; else hi[c - 8] = (_Float16)p;
      }
      *(v8h*)(&attW[wid][r][sub * 16])     = lo;
      *(v8h*)(&attW[wid][r][sub * 16 + 8]) = hi;
      v16h af  = ld_a_frag(&attW[wid][r][0], sub);
      v16h bf0 = ld_b_frag(&hT[r][kc * 32], sub);
      v16h bf1 = ld_b_frag(&hT[16 + r][kc * 32], sub);
      acc0 = wmma_f16(af, bf0, acc0);
      acc1 = wmma_f16(af, bf1, acc1);
    }
    psum += __shfl_xor(psum, 16, 32);
    if (sub == 0) invdW[wid][r] = (psum > 0.f) ? 1.f / psum : 0.f;

    // epilogue: normalize, bnh -> elu -> bnt, write Xcat
    for (int rr = 0; rr < 8; ++rr) {
      int mrow = rr + 8 * sub;
      int orow = row0 + mrow;
      if (orow < NN) {
        float inv = invdW[wid][mrow];
        float g1 = bnh_g[h * NN + orow], bb1 = bnh_b[h * NN + orow];
        float g2 = bnt_g[orow],          bb2 = bnt_b[orow];
        float v0 = acc0[rr] * inv; v0 = v0 * BNS * g1 + bb1; v0 = eluf(v0); v0 = v0 * BNS * g2 + bb2;
        float v1 = acc1[rr] * inv; v1 = v1 * BNS * g1 + bb1; v1 = eluf(v1); v1 = v1 * BNS * g2 + bb2;
        xcat[((size_t)b * NN + orow) * HO + h * FO + r]      = v0;
        xcat[((size_t)b * NN + orow) * HO + h * FO + 16 + r] = v1;
      }
    }
  }
}

// ---------------- kernel 4: second-level projections + scores (fp32 VALU, K=256) ----------------
__global__ void k4_proj2(const float* __restrict__ xcat,
                         const float* __restrict__ Ws, const float* __restrict__ as_,
                         const float* __restrict__ Wp, const float* __restrict__ ap_,
                         const float* __restrict__ Wq, const float* __restrict__ aq_,
                         float* __restrict__ Hq, float* __restrict__ Hs, float* __restrict__ Hp2,
                         float* __restrict__ s1s, float* __restrict__ s2s,
                         float* __restrict__ s1p, float* __restrict__ s2p,
                         float* __restrict__ s1q, float* __restrict__ s2q) {
  int idx = blockIdx.x * 256 + threadIdx.x;
  if (idx >= BB * NN) return;
  const float* xr = xcat + (size_t)idx * HO;
  float hq[FO];
#pragma unroll
  for (int i = 0; i < FO; ++i) hq[i] = 0.f;
  float hs0 = 0.f, hs1 = 0.f, hp0 = 0.f, hp1 = 0.f;
  for (int k = 0; k < HO; ++k) {
    float xv = xr[k];
    const float* wq = Wq + k * FO;
#pragma unroll
    for (int fo = 0; fo < FO; ++fo) hq[fo] += xv * wq[fo];
    hs0 += xv * Ws[k * 2]; hs1 += xv * Ws[k * 2 + 1];
    hp0 += xv * Wp[k * 2]; hp1 += xv * Wp[k * 2 + 1];
  }
  float x1q = 0.f, x2q = 0.f;
#pragma unroll
  for (int fo = 0; fo < FO; ++fo) {
    Hq[(size_t)idx * FO + fo] = hq[fo];
    x1q += hq[fo] * aq_[fo];
    x2q += hq[fo] * aq_[FO + fo];
  }
  Hs[idx * 2] = hs0;  Hs[idx * 2 + 1] = hs1;
  Hp2[idx * 2] = hp0; Hp2[idx * 2 + 1] = hp1;
  s1s[idx] = hs0 * as_[0] + hs1 * as_[1];
  s2s[idx] = hs0 * as_[2] + hs1 * as_[3];
  s1p[idx] = hp0 * ap_[0] + hp1 * ap_[1];
  s2p[idx] = hp0 * ap_[2] + hp1 * ap_[3];
  s1q[idx] = x1q;
  s2q[idx] = x2q;
}

// ---------------- kernel 5: second-level attention (sent sigmoid, para elu, Qtype) ----------------
__global__ void k5_attn2(const int* __restrict__ adj,
                         const float* __restrict__ s1s, const float* __restrict__ s2s,
                         const float* __restrict__ s1p, const float* __restrict__ s2p,
                         const float* __restrict__ s1q, const float* __restrict__ s2q,
                         const float* __restrict__ Hs, const float* __restrict__ Hp2,
                         const float* __restrict__ Hq,
                         const float* __restrict__ bns_g, const float* __restrict__ bns_b,
                         const float* __restrict__ bnp_g, const float* __restrict__ bnp_b,
                         const float* __restrict__ bnq_g, const float* __restrict__ bnq_b,
                         const float* __restrict__ W2, float* __restrict__ out) {
  __shared__ float S1s[NN], S2s[NN], S1p[NN], S2p[NN], S2q[NN];
  __shared__ float HsL[NN][2], HpL[NN][2];
  __shared__ float eQ[NN], pQ[NN], qVec[FO], redQ[2];
  const int b = blockIdx.x, tid = threadIdx.x;
  for (int i = tid; i < NN; i += 256) {
    int g = b * NN + i;
    S1s[i] = s1s[g]; S2s[i] = s2s[g];
    S1p[i] = s1p[g]; S2p[i] = s2p[g];
    S2q[i] = s2q[g];
    HsL[i][0] = Hs[g * 2]; HsL[i][1] = Hs[g * 2 + 1];
    HpL[i][0] = Hp2[g * 2]; HpL[i][1] = Hp2[g * 2 + 1];
  }
  __syncthreads();

  if (tid < NN) {
    const int i = tid;
    const int* adjR = adj + ((size_t)b * NN + i) * NN;
    // sentence head (sigmoid)
    {
      float mx = -3.0e38f;
      for (int j = 0; j < NN; ++j) {
        float e = leakyf(S1s[i] + S2s[j]);
        mx = fmaxf(mx, (adjR[j] > 0) ? e : NEGV);
      }
      float sum = 0.f, a0 = 0.f, a1 = 0.f;
      for (int j = 0; j < NN; ++j) {
        float e  = leakyf(S1s[i] + S2s[j]);
        float p  = __expf(((adjR[j] > 0) ? e : NEGV) - mx);
        sum += p; a0 += p * HsL[j][0]; a1 += p * HsL[j][1];
      }
      float inv = 1.f / sum;
      float v0 = a0 * inv * BNS * bns_g[i] + bns_b[i];
      float v1 = a1 * inv * BNS * bns_g[i] + bns_b[i];
      out[((size_t)b * NN + i) * 2]     = 1.f / (1.f + __expf(-v0));
      out[((size_t)b * NN + i) * 2 + 1] = 1.f / (1.f + __expf(-v1));
    }
    // paragraph head (elu)
    {
      float mx = -3.0e38f;
      for (int j = 0; j < NN; ++j) {
        float e = leakyf(S1p[i] + S2p[j]);
        mx = fmaxf(mx, (adjR[j] > 0) ? e : NEGV);
      }
      float sum = 0.f, a0 = 0.f, a1 = 0.f;
      for (int j = 0; j < NN; ++j) {
        float e  = leakyf(S1p[i] + S2p[j]);
        float p  = __expf(((adjR[j] > 0) ? e : NEGV) - mx);
        sum += p; a0 += p * HpL[j][0]; a1 += p * HpL[j][1];
      }
      float inv = 1.f / sum;
      float v0 = a0 * inv * BNS * bnp_g[i] + bnp_b[i];
      float v1 = a1 * inv * BNS * bnp_g[i] + bnp_b[i];
      size_t off = (size_t)BB * NN * 2;
      out[off + ((size_t)b * NN + i) * 2]     = eluf(v0);
      out[off + ((size_t)b * NN + i) * 2 + 1] = eluf(v1);
    }
  }

  // query head: only row 0 of attention output is needed
  {
    float s1q0 = s1q[b * NN];
    const int* adj0 = adj + (size_t)b * NN * NN;
    if (tid < NN) {
      float e = leakyf(s1q0 + S2q[tid]);
      eQ[tid] = (adj0[tid] > 0) ? e : NEGV;
    }
    __syncthreads();
    if (tid == 0) {
      float mx = -3.0e38f;
      for (int j = 0; j < NN; ++j) mx = fmaxf(mx, eQ[j]);
      redQ[0] = mx;
    }
    __syncthreads();
    if (tid < NN) pQ[tid] = __expf(eQ[tid] - redQ[0]);
    __syncthreads();
    if (tid == 0) {
      float sum = 0.f;
      for (int j = 0; j < NN; ++j) sum += pQ[j];
      redQ[1] = 1.f / sum;
    }
    __syncthreads();
    if (tid < FO) {
      float acc = 0.f;
      for (int j = 0; j < NN; ++j) acc += pQ[j] * Hq[((size_t)b * NN + j) * FO + tid];
      qVec[tid] = acc * redQ[1] * BNS * bnq_g[0] + bnq_b[0];
    }
    __syncthreads();
    if (tid < 2) {
      float v = 0.f;
      for (int fo = 0; fo < FO; ++fo) v += qVec[fo] * W2[fo * 2 + tid];
      out[(size_t)BB * NN * 4 + b * 2 + tid] = eluf(v);
    }
  }
}

extern "C" void kernel_launch(void* const* d_in, const int* in_sizes, int n_in,
                              void* d_out, int out_size, void* d_ws, size_t ws_size,
                              hipStream_t stream) {
  const float* feat   = (const float*)d_in[0];
  const int*   adj    = (const int*)d_in[1];
  const float* Wheads = (const float*)d_in[2];
  const float* aheads = (const float*)d_in[3];
  const float* bnh_g  = (const float*)d_in[4];
  const float* bnh_b  = (const float*)d_in[5];
  const float* bnt_g  = (const float*)d_in[6];
  const float* bnt_b  = (const float*)d_in[7];
  const float* Wsent  = (const float*)d_in[8];
  const float* asent  = (const float*)d_in[9];
  const float* bns_g  = (const float*)d_in[10];
  const float* bns_b  = (const float*)d_in[11];
  const float* Wpara  = (const float*)d_in[12];
  const float* apara  = (const float*)d_in[13];
  const float* bnp_g  = (const float*)d_in[14];
  const float* bnp_b  = (const float*)d_in[15];
  const float* Wq     = (const float*)d_in[16];
  const float* aq     = (const float*)d_in[17];
  const float* bnq_g  = (const float*)d_in[18];
  const float* bnq_b  = (const float*)d_in[19];
  const float* W2     = (const float*)d_in[20];
  float* out = (float*)d_out;

  char* ws = (char*)d_ws;
  size_t off = 0;
  auto alloc = [&](size_t bytes) { void* p = ws + off; off = (off + bytes + 255) & ~(size_t)255; return p; };
  _Float16* wallT = (_Float16*)alloc((size_t)HO * FF * 2);
  _Float16* featH = (_Float16*)alloc((size_t)BB * NP * FF * 2);
  float* hall = (float*)alloc((size_t)BB * NN * HO * 4);
  float* s1h  = (float*)alloc((size_t)HH * BB * NN * 4);
  float* s2h  = (float*)alloc((size_t)HH * BB * NN * 4);
  float* xcat = (float*)alloc((size_t)BB * NN * HO * 4);
  float* Hq   = (float*)alloc((size_t)BB * NN * FO * 4);
  float* Hs   = (float*)alloc((size_t)BB * NN * 2 * 4);
  float* Hp2  = (float*)alloc((size_t)BB * NN * 2 * 4);
  float* s1s  = (float*)alloc((size_t)BB * NN * 4);
  float* s2s  = (float*)alloc((size_t)BB * NN * 4);
  float* s1p  = (float*)alloc((size_t)BB * NN * 4);
  float* s2p  = (float*)alloc((size_t)BB * NN * 4);
  float* s1q  = (float*)alloc((size_t)BB * NN * 4);
  float* s2q  = (float*)alloc((size_t)BB * NN * 4);

  k0_prep_w<<<(HO * FF + 255) / 256, 256, 0, stream>>>(Wheads, wallT);
  k0_prep_x<<<(BB * NP * FF / 8 + 255) / 256, 256, 0, stream>>>(feat, featH);
  k1_gemm<<<dim3(13, BB), 256, 0, stream>>>(featH, wallT, hall);
  k2_scores<<<(HH * BB * NN + 255) / 256, 256, 0, stream>>>(hall, aheads, s1h, s2h);
  k3_attn<<<dim3(HH, BB), 256, 0, stream>>>(hall, s1h, s2h, adj, bnh_g, bnh_b, bnt_g, bnt_b, xcat);
  k4_proj2<<<(BB * NN + 255) / 256, 256, 0, stream>>>(xcat, Wsent, asent, Wpara, apara, Wq, aq,
                                                      Hq, Hs, Hp2, s1s, s2s, s1p, s2p, s1q, s2q);
  k5_attn2<<<BB, 256, 0, stream>>>(adj, s1s, s2s, s1p, s2p, s1q, s2q, Hs, Hp2, Hq,
                                   bns_g, bns_b, bnp_g, bnp_b, bnq_g, bnq_b, W2, out);
}
